// DeepseekV3Attention_34961033789977
// MI455X (gfx1250) — matrix operand order repaired and checked by reading, not compiled, not run on hardware
//
#include <hip/hip_runtime.h>

typedef __bf16 bf16;
typedef __attribute__((ext_vector_type(16))) __bf16 v16bf;
typedef __attribute__((ext_vector_type(8)))  __bf16 v8bf;
typedef __attribute__((ext_vector_type(2)))  __bf16 v2bf;
typedef __attribute__((ext_vector_type(8)))  float  v8f;
typedef __attribute__((ext_vector_type(4)))  int    v4i;

#define AS1 __attribute__((address_space(1)))
#define AS3 __attribute__((address_space(3)))

#define B_    2
#define S_    2048
#define D_    2048
#define H_    16
#define NOPE_ 128
#define ROPE_ 64
#define VDIM_ 128
#define QHD_  192
#define QR_   1536
#define KVR_  512
#define BS_   (B_*S_)
#define EPS_  1e-6f
#define SCALE_ 0.0721687836487032f   /* 192^-0.5 */

#ifndef __has_builtin
#define __has_builtin(x) 0
#endif
#if __has_builtin(__builtin_amdgcn_global_load_async_to_lds_b128)
#define HAVE_ASYNC_LDS 1
#else
#define HAVE_ASYNC_LDS 0
#endif

// ---------------------------------------------------------------------------
// helpers
// ---------------------------------------------------------------------------
__device__ __forceinline__ v8f v8f_zero() {
  v8f z;
  for (int i = 0; i < 8; ++i) z[i] = 0.f;
  return z;
}

__device__ __forceinline__ void lds_fence() {
  asm volatile("s_wait_dscnt 0" ::: "memory");
}

__device__ __forceinline__ void async_fence() {
  asm volatile("s_wait_asynccnt 0" ::: "memory");
}

// A-operand fragment (16x32 bf16, ISA 16-bit A layout):
// lanes 0-15 = rows M=0..15 holding K 0..7 and 16..23; lanes 16-31 hold K 8..15, 24..31
__device__ __forceinline__ v16bf frag_a(const bf16* tile, int ld) {
  int lane = threadIdx.x & 31;
  const bf16* r = tile + (size_t)(lane & 15) * ld + ((lane >> 4) & 1) * 8;
  v16bf f;
  *reinterpret_cast<v8bf*>(&f)       = *reinterpret_cast<const v8bf*>(r);
  *(reinterpret_cast<v8bf*>(&f) + 1) = *reinterpret_cast<const v8bf*>(r + 16);
  return f;
}

// B-operand fragment (32x16 bf16): lanes 0-15 = col N holding K 0..7 and
// 16..23, lanes 16-31 hold K 8..15, 24..31. `tile` is column-major
// (row = N, ld elements of K per row).
__device__ __forceinline__ v16bf frag_b(const bf16* tile, int ld) {
  int lane = threadIdx.x & 31;
  const bf16* r = tile + (size_t)(lane & 15) * ld + ((lane >> 4) & 1) * 8;
  v16bf f;
  *reinterpret_cast<v8bf*>(&f)       = *reinterpret_cast<const v8bf*>(r);
  *(reinterpret_cast<v8bf*>(&f) + 1) = *reinterpret_cast<const v8bf*>(r + 16);
  return f;
}

__device__ __forceinline__ v8f wmma_bf16(v16bf a, v16bf b, v8f c) {
  return __builtin_amdgcn_wmma_f32_16x16x32_bf16(false, a, false, b, (short)0, c,
                                                 false, false);
}

// ---------------------------------------------------------------------------
// fp32 -> bf16 conversion
// ---------------------------------------------------------------------------
__global__ __launch_bounds__(256) void cvt_f32_bf16(const float* __restrict__ src,
                                                    bf16* __restrict__ dst, long n) {
  long i = (long)blockIdx.x * 256 + threadIdx.x;
  long stride = (long)gridDim.x * 256;
  for (; i < n; i += stride) dst[i] = (bf16)src[i];
}

// ---------------------------------------------------------------------------
// RMS norm: src fp32 [rows, srcStride] (first n used), dst bf16 [rows, n]
// ---------------------------------------------------------------------------
__global__ __launch_bounds__(256) void rmsnorm_kernel(const float* __restrict__ src,
                                                      const float* __restrict__ w,
                                                      bf16* __restrict__ dst,
                                                      int n, int srcStride) {
  int row = blockIdx.x;
  const float* x = src + (size_t)row * srcStride;
  float ss = 0.f;
  for (int i = threadIdx.x; i < n; i += 256) {
    float v = x[i];
    ss += v * v;
  }
  for (int off = 16; off > 0; off >>= 1) ss += __shfl_xor(ss, off);
  __shared__ float red[8];
  __shared__ float s_rstd;
  if ((threadIdx.x & 31) == 0) red[threadIdx.x >> 5] = ss;
  __syncthreads();
  if (threadIdx.x == 0) {
    float tot = 0.f;
    for (int i = 0; i < 8; ++i) tot += red[i];
    s_rstd = rsqrtf(tot / (float)n + EPS_);
  }
  __syncthreads();
  float rs = s_rstd;
  for (int i = threadIdx.x; i < n; i += 256)
    dst[(size_t)row * n + i] = (bf16)(w[i] * x[i] * rs);
}

// ---------------------------------------------------------------------------
// bf16 GEMM: C[M,N] f32 = A[M,K] * B[K,N]  (row-major)
// Requirements: M % 128 == 0, K % 32 == 0, N % 8 == 0.
// 128x128x32 tiles, 8 waves, double-buffered LDS, software pipelined.
// A tile staged via async global->LDS copies; B staged transposed (packed b32).
// ---------------------------------------------------------------------------
#define GBM 128
#define GBN 128
#define GBK 32
#define GLD 40   // padded LDS row stride (elements); 80B keeps 16B alignment

__global__ __launch_bounds__(256) void gemm_bf16(const bf16* __restrict__ A,
                                                 const bf16* __restrict__ B,
                                                 float* __restrict__ C,
                                                 int M, int N, int K) {
  __shared__ bf16 As[2][GBM * GLD];
  __shared__ bf16 BsT[2][GBN * GLD];   // transposed: [n][k]

  int m0 = blockIdx.y * GBM, n0 = blockIdx.x * GBN;
  int t = threadIdx.x;
  int lane = t & 31;
  int wave = t >> 5;
  int wm = wave >> 2;      // 0..1
  int wn = wave & 3;       // 0..3

  // A staging: 512 chunks of 8 bf16 (128 rows x 4 segs), 2 per thread
  int aRow[2], aSeg[2];
  for (int i = 0; i < 2; ++i) {
    int c = t + i * 256;
    aRow[i] = c >> 2;
    aSeg[i] = c & 3;
  }
  // B staging: thread owns k-pair (bK0, bK0+1) x 8 columns -> packed b32 stores
  int bK0 = (t >> 4) * 2;      // even k in 0..30
  int bN8 = t & 15;            // column octet

  v8bf aReg[2], bReg[2];

  auto a_gload = [&](int kk) {
    for (int i = 0; i < 2; ++i)
      aReg[i] = *reinterpret_cast<const v8bf*>(
          A + (size_t)(m0 + aRow[i]) * K + kk + aSeg[i] * 8);
  };
  auto a_store = [&](int p) {
    for (int i = 0; i < 2; ++i)
      *reinterpret_cast<v8bf*>(&As[p][aRow[i] * GLD + aSeg[i] * 8]) = aReg[i];
  };
  auto b_gload = [&](int kk) {
    int gn = n0 + bN8 * 8;
    if (gn + 7 < N) {
      bReg[0] = *reinterpret_cast<const v8bf*>(B + (size_t)(kk + bK0) * N + gn);
      bReg[1] = *reinterpret_cast<const v8bf*>(B + (size_t)(kk + bK0 + 1) * N + gn);
    } else {
      for (int e = 0; e < 8; ++e) { bReg[0][e] = (bf16)0.f; bReg[1][e] = (bf16)0.f; }
    }
  };
  auto b_store = [&](int p) {
    for (int e = 0; e < 8; ++e) {
      v2bf pk;
      pk[0] = bReg[0][e];
      pk[1] = bReg[1][e];
      *reinterpret_cast<v2bf*>(&BsT[p][(bN8 * 8 + e) * GLD + bK0]) = pk;
    }
  };
#if HAVE_ASYNC_LDS
  auto a_async = [&](int kk, int p) {
    for (int i = 0; i < 2; ++i)
      __builtin_amdgcn_global_load_async_to_lds_b128(
          (AS1 v4i*)(A + (size_t)(m0 + aRow[i]) * K + kk + aSeg[i] * 8),
          (AS3 v4i*)(&As[p][aRow[i] * GLD + aSeg[i] * 8]),
          0, 0);
  };
#endif

  v8f acc[4][2];
  for (int mi = 0; mi < 4; ++mi)
    for (int ni = 0; ni < 2; ++ni) acc[mi][ni] = v8f_zero();

  // prologue: stage tile 0 into buffer 0
#if HAVE_ASYNC_LDS
  a_async(0, 0);
  b_gload(0);
  b_store(0);
  async_fence();
#else
  a_gload(0);
  b_gload(0);
  a_store(0);
  b_store(0);
#endif
  __syncthreads();

  int p = 0;
  for (int kk = 0; kk < K; kk += GBK) {
    bool more = (kk + GBK) < K;
    // prefetch next tile while computing this one
#if HAVE_ASYNC_LDS
    if (more) { a_async(kk + GBK, p ^ 1); b_gload(kk + GBK); }
#else
    if (more) { a_gload(kk + GBK); b_gload(kk + GBK); }
#endif

    v16bf bfrag[2];
    for (int ni = 0; ni < 2; ++ni)
      bfrag[ni] = frag_b(&BsT[p][(wn * 32 + ni * 16) * GLD], GLD);
    for (int mi = 0; mi < 4; ++mi) {
      v16bf afrag = frag_a(&As[p][(wm * 64 + mi * 16) * GLD], GLD);
      for (int ni = 0; ni < 2; ++ni)
        acc[mi][ni] = wmma_bf16(afrag, bfrag[ni], acc[mi][ni]);
    }

#if HAVE_ASYNC_LDS
    if (more) b_store(p ^ 1);
    async_fence();
#else
    if (more) { a_store(p ^ 1); b_store(p ^ 1); }
#endif
    __syncthreads();
    p ^= 1;
  }

  int lhalf = (lane >> 4) & 1, lcol = lane & 15;
  for (int mi = 0; mi < 4; ++mi)
    for (int ni = 0; ni < 2; ++ni)
      for (int r = 0; r < 8; ++r) {
        int row = m0 + wm * 64 + mi * 16 + r + lhalf * 8;   // M % 128 == 0
        int col = n0 + wn * 32 + ni * 16 + lcol;
        if (col < N) C[(size_t)row * N + col] = acc[mi][ni][r];
      }
}

// ---------------------------------------------------------------------------
// RoPE helpers + head assembly
// ---------------------------------------------------------------------------
__device__ __forceinline__ float rope_val(const float* base, int j, int s) {
  float x = base[j];
  int jp = (j < 32) ? j + 32 : j - 32;
  float rh = (j < 32) ? -base[jp] : base[jp];
  float inv = __powf(10000.f, -(float)(2 * (j & 31)) / 64.f);
  float ang = (float)s * inv;
  return x * __cosf(ang) + rh * __sinf(ang);
}

// q fp32 [B*S, H*192] -> Qh bf16 [B,H,S,192] with RoPE on last 64 dims
__global__ __launch_bounds__(256) void build_q_kernel(const float* __restrict__ q,
                                                      bf16* __restrict__ Qh) {
  int bs = blockIdx.x;
  int b = bs / S_, s = bs % S_;
  const float* qr = q + (size_t)bs * (H_ * QHD_);
  for (int i = threadIdx.x; i < H_ * QHD_; i += 256) {
    int h = i / QHD_, d = i % QHD_;
    float v;
    if (d < NOPE_) v = qr[i];
    else           v = rope_val(qr + h * QHD_ + NOPE_, d - NOPE_, s);
    Qh[(((size_t)(b * H_ + h)) * S_ + s) * QHD_ + d] = (bf16)v;
  }
}

// kvf fp32 [B*S, H*256], ckv fp32 [B*S, 576] -> Kh [B,H,S,192], Vh [B,H,S,128]
__global__ __launch_bounds__(256) void build_kv_kernel(const float* __restrict__ kvf,
                                                       const float* __restrict__ ckv,
                                                       bf16* __restrict__ Kh,
                                                       bf16* __restrict__ Vh) {
  int bs = blockIdx.x;
  int b = bs / S_, s = bs % S_;
  const float* kr = kvf + (size_t)bs * (H_ * (NOPE_ + VDIM_));
  const float* cr = ckv + (size_t)bs * (KVR_ + ROPE_);
  __shared__ float pe[ROPE_];
  if (threadIdx.x < ROPE_) pe[threadIdx.x] = rope_val(cr + KVR_, threadIdx.x, s);
  __syncthreads();
  for (int i = threadIdx.x; i < H_ * QHD_; i += 256) {
    int h = i / QHD_, d = i % QHD_;
    float v = (d < NOPE_) ? kr[h * (NOPE_ + VDIM_) + d] : pe[d - NOPE_];
    Kh[(((size_t)(b * H_ + h)) * S_ + s) * QHD_ + d] = (bf16)v;
  }
  for (int i = threadIdx.x; i < H_ * VDIM_; i += 256) {
    int h = i / VDIM_, d = i % VDIM_;
    Vh[(((size_t)(b * H_ + h)) * S_ + s) * VDIM_ + d] =
        (bf16)kr[h * (NOPE_ + VDIM_) + NOPE_ + d];
  }
}

// ---------------------------------------------------------------------------
// Flash attention: Qh/Kh [B,H,S,192], Vh [B,H,S,128] -> Attn bf16 [B,S,H*128]
// Block = 4 waves cooperating on one 64-row q tile; K/V tiles staged once per
// 32-k step into shared double-buffered LDS (K via async global->LDS copies).
// Uniform trip count across waves; one __syncthreads per step.
// ---------------------------------------------------------------------------
#define KLD 200   // K tile LDS row stride (192 + 8 pad), 400B rows (16B aligned)
#define FLD 40
__global__ __launch_bounds__(128) void flash_kernel(const bf16* __restrict__ Qh,
                                                    const bf16* __restrict__ Kh,
                                                    const bf16* __restrict__ Vh,
                                                    bf16* __restrict__ Attn) {
  __shared__ bf16 Ks[2][32 * KLD];     // K tile  [32 krow][192 d]  row-major
  __shared__ bf16 VsT[2][128 * FLD];   // V tile transposed [128 d][32 k]
  __shared__ bf16 Ps[4][16 * FLD];     // per-wave P tile [16 q][32 k]

  int bh = blockIdx.y;             // b*H + h
  int b = bh / H_, h = bh % H_;
  int t = threadIdx.x;
  int wave = t >> 5, lane = t & 31;
  int lhalf = (lane >> 4) & 1, lcol = lane & 15;
  int qbase = blockIdx.x * 64;
  int q0 = qbase + wave * 16;
  int kmax = qbase + 64;           // uniform causal bound for whole block

  const bf16* Qb = Qh + ((size_t)bh * S_ + q0) * QHD_;
  const bf16* Kb = Kh + (size_t)bh * S_ * QHD_;
  const bf16* Vb = Vh + (size_t)bh * S_ * VDIM_;

  // V staging: 256 tasks (16 k-pairs x 16 d-octets), 2 per thread
  v8bf vReg[2][2];
  auto v_gload = [&](int k0) {
    for (int i = 0; i < 2; ++i) {
      int c = t + i * 128;
      int kp = c >> 4, d8 = c & 15;
      vReg[i][0] = *reinterpret_cast<const v8bf*>(
          Vb + (size_t)(k0 + 2 * kp) * VDIM_ + d8 * 8);
      vReg[i][1] = *reinterpret_cast<const v8bf*>(
          Vb + (size_t)(k0 + 2 * kp + 1) * VDIM_ + d8 * 8);
    }
  };
  auto v_store = [&](int p) {
    for (int i = 0; i < 2; ++i) {
      int c = t + i * 128;
      int kp = c >> 4, d8 = c & 15;
      for (int e = 0; e < 8; ++e) {
        v2bf pk;
        pk[0] = vReg[i][0][e];
        pk[1] = vReg[i][1][e];
        *reinterpret_cast<v2bf*>(&VsT[p][(d8 * 8 + e) * FLD + 2 * kp]) = pk;
      }
    }
  };
  // K staging: 32 rows x 24 b128 chunks = 768 chunks, 6 per thread
#if HAVE_ASYNC_LDS
  auto k_stage = [&](int k0, int p) {
    for (int i = 0; i < 6; ++i) {
      int c = t + i * 128;
      int row = c / 24, seg = c % 24;
      __builtin_amdgcn_global_load_async_to_lds_b128(
          (AS1 v4i*)(Kb + (size_t)(k0 + row) * QHD_ + seg * 8),
          (AS3 v4i*)(&Ks[p][row * KLD + seg * 8]), 0, 0);
    }
  };
#else
  auto k_stage = [&](int k0, int p) {
    for (int i = 0; i < 6; ++i) {
      int c = t + i * 128;
      int row = c / 24, seg = c % 24;
      *reinterpret_cast<v8bf*>(&Ks[p][row * KLD + seg * 8]) =
          *reinterpret_cast<const v8bf*>(Kb + (size_t)(k0 + row) * QHD_ + seg * 8);
    }
  };
#endif

  v16bf qf[6];
  for (int c = 0; c < 6; ++c) qf[c] = frag_a(Qb + c * 32, QHD_);

  v8f acc[8];
  for (int i = 0; i < 8; ++i) acc[i] = v8f_zero();
  float mrow[8], lrow[8];
  for (int r = 0; r < 8; ++r) { mrow[r] = -1e30f; lrow[r] = 0.f; }

  // prologue: stage step 0 into buffer 0
  k_stage(0, 0);
  v_gload(0);
  v_store(0);
  async_fence();
  __syncthreads();

  int p = 0;
  for (int k0 = 0; k0 < kmax; k0 += 32) {
    bool more = (k0 + 32) < kmax;
    if (more) { k_stage(k0 + 32, p ^ 1); v_gload(k0 + 32); }

    // ---- scores S[16 q, 32 k] = Q . K^T from LDS K tile ----
    v8f sc[2];
    for (int half = 0; half < 2; ++half) {
      sc[half] = v8f_zero();
      for (int c = 0; c < 6; ++c) {
        v16bf kfrag = frag_b(&Ks[p][(half * 16) * KLD + c * 32], KLD);
        sc[half] = wmma_bf16(qf[c], kfrag, sc[half]);
      }
    }

    // ---- scale + causal mask + online softmax ----
    for (int r = 0; r < 8; ++r) {
      int grow = q0 + r + lhalf * 8;
      int c0 = k0 + lcol, c1 = k0 + 16 + lcol;
      float a0 = sc[0][r] * SCALE_ + ((c0 > grow) ? -1e9f : 0.f);
      float a1 = sc[1][r] * SCALE_ + ((c1 > grow) ? -1e9f : 0.f);
      float vmax = fmaxf(a0, a1);
      for (int off = 8; off > 0; off >>= 1) vmax = fmaxf(vmax, __shfl_xor(vmax, off));
      float m_new = fmaxf(mrow[r], vmax);
      float p0 = __expf(a0 - m_new);
      float p1 = __expf(a1 - m_new);
      float rsum = p0 + p1;
      for (int off = 8; off > 0; off >>= 1) rsum += __shfl_xor(rsum, off);
      float corr = __expf(mrow[r] - m_new);
      lrow[r] = lrow[r] * corr + rsum;
      mrow[r] = m_new;
      for (int i = 0; i < 8; ++i) acc[i][r] *= corr;
      int M = r + lhalf * 8;
      Ps[wave][M * FLD + lcol]      = (bf16)p0;
      Ps[wave][M * FLD + 16 + lcol] = (bf16)p1;
    }
    lds_fence();

    // ---- O += P . V from LDS ----
    v16bf pf = frag_a(&Ps[wave][0], FLD);
    for (int i = 0; i < 8; ++i) {
      v16bf vf = frag_b(&VsT[p][i * 16 * FLD], FLD);
      acc[i] = wmma_bf16(pf, vf, acc[i]);
    }

    if (more) v_store(p ^ 1);
    async_fence();
    __syncthreads();
    p ^= 1;
  }

  // ---- normalize + store to [B,S,H*VDIM] ----
  float linv[8];
  for (int r = 0; r < 8; ++r) linv[r] = 1.f / lrow[r];
  bf16* Ab = Attn + ((size_t)b * S_ + q0) * (H_ * VDIM_) + h * VDIM_;
  for (int i = 0; i < 8; ++i)
    for (int r = 0; r < 8; ++r) {
      int M = r + lhalf * 8;
      Ab[(size_t)M * (H_ * VDIM_) + i * 16 + lcol] = (bf16)(acc[i][r] * linv[r]);
    }
}

// ---------------------------------------------------------------------------
// launch
// ---------------------------------------------------------------------------
extern "C" void kernel_launch(void* const* d_in, const int* in_sizes, int n_in,
                              void* d_out, int out_size, void* d_ws, size_t ws_size,
                              hipStream_t stream) {
  (void)in_sizes; (void)n_in; (void)out_size; (void)ws_size;
  const float* hidden = (const float*)d_in[0];
  const float* w_qa   = (const float*)d_in[3];
  const float* qa_ln  = (const float*)d_in[4];
  const float* w_qb   = (const float*)d_in[5];
  const float* w_kva  = (const float*)d_in[6];
  const float* kva_ln = (const float*)d_in[7];
  const float* w_kvb  = (const float*)d_in[8];
  const float* w_o    = (const float*)d_in[9];
  float* out = (float*)d_out;

  char* ws = (char*)d_ws;
  size_t off = 0;
  auto take = [&](size_t bytes) -> char* {
    char* p = ws + off;
    off = (off + bytes + 255) & ~(size_t)255;
    return p;
  };
  bf16* Xb   = (bf16*)take((size_t)BS_ * D_ * 2);            // hidden bf16
  bf16* Wbuf = (bf16*)take((size_t)QR_ * H_ * QHD_ * 2);     // largest weight (w_qb)
  char* arena = take(88080384);                              // overlapped Q / KV scratch
  bf16* Qh   = (bf16*)take((size_t)B_ * H_ * S_ * QHD_ * 2);
  bf16* Kh   = (bf16*)take((size_t)B_ * H_ * S_ * QHD_ * 2);
  bf16* Vh   = (bf16*)take((size_t)B_ * H_ * S_ * VDIM_ * 2);
  bf16* Attn = (bf16*)take((size_t)BS_ * H_ * VDIM_ * 2);

  // Q-path arena layout
  float* QA  = (float*)(arena);                         // [4096,1536] f32
  bf16*  QAn = (bf16*)(arena + 25165824);               // [4096,1536] bf16
  float* Qf  = (float*)(arena + 37748736);              // [4096,3072] f32
  // KV-path arena layout (reuses same arena after Q path is consumed)
  float* CKV = (float*)(arena);                         // [4096,576] f32
  bf16*  KVn = (bf16*)(arena + 9437184);                // [4096,512] bf16
  float* KVf = (float*)(arena + 13631488);              // [4096,4096] f32

  dim3 blk256(256), blk128(128);

  cvt_f32_bf16<<<2048, blk256, 0, stream>>>(hidden, Xb, (long)BS_ * D_);

  // --- Q path ---
  cvt_f32_bf16<<<2048, blk256, 0, stream>>>(w_qa, Wbuf, (long)D_ * QR_);
  gemm_bf16<<<dim3(QR_ / 128, BS_ / 128), blk256, 0, stream>>>(Xb, Wbuf, QA, BS_, QR_, D_);
  rmsnorm_kernel<<<BS_, blk256, 0, stream>>>(QA, qa_ln, QAn, QR_, QR_);
  cvt_f32_bf16<<<2048, blk256, 0, stream>>>(w_qb, Wbuf, (long)QR_ * H_ * QHD_);
  gemm_bf16<<<dim3(H_ * QHD_ / 128, BS_ / 128), blk256, 0, stream>>>(QAn, Wbuf, Qf, BS_, H_ * QHD_, QR_);
  build_q_kernel<<<BS_, blk256, 0, stream>>>(Qf, Qh);

  // --- KV path (arena reuse) ---
  cvt_f32_bf16<<<2048, blk256, 0, stream>>>(w_kva, Wbuf, (long)D_ * (KVR_ + ROPE_));
  gemm_bf16<<<dim3((KVR_ + ROPE_ + 127) / 128, BS_ / 128), blk256, 0, stream>>>(
      Xb, Wbuf, CKV, BS_, KVR_ + ROPE_, D_);
  rmsnorm_kernel<<<BS_, blk256, 0, stream>>>(CKV, kva_ln, KVn, KVR_, KVR_ + ROPE_);
  cvt_f32_bf16<<<2048, blk256, 0, stream>>>(w_kvb, Wbuf, (long)KVR_ * H_ * (NOPE_ + VDIM_));
  gemm_bf16<<<dim3(H_ * (NOPE_ + VDIM_) / 128, BS_ / 128), blk256, 0, stream>>>(
      KVn, Wbuf, KVf, BS_, H_ * (NOPE_ + VDIM_), KVR_);
  build_kv_kernel<<<BS_, blk256, 0, stream>>>(KVf, CKV, Kh, Vh);

  // --- attention ---
  flash_kernel<<<dim3(S_ / 64, B_ * H_), blk128, 0, stream>>>(Qh, Kh, Vh, Attn);

  // --- output projection ---
  cvt_f32_bf16<<<2048, blk256, 0, stream>>>(w_o, Wbuf, (long)H_ * VDIM_ * D_);
  gemm_bf16<<<dim3(D_ / 128, BS_ / 128), blk256, 0, stream>>>(Attn, Wbuf, out, BS_, D_, H_ * VDIM_);
}
